// LSTMPeepholeCell_55173149884860
// MI455X (gfx1250) — compile-verified, matrix operand order checked by reading, not run on hardware
//
#include <hip/hip_runtime.h>
#include <stdint.h>

// ---------------------------------------------------------------------------
// Fused peephole-LSTM cell for MI455X (gfx1250, wave32, WMMA).
// Pass 1 (prep): X,H -> XH bf16 [NB,1024]; W,R -> Wt bf16 [NZ,1024] (z-col major)
// Pass 2 (fused): z = XH @ Wt^T + b via v_wmma_f32_16x16x32_bf16, f32 accum,
//                 gate/peephole/h/c tail fused. Tiles staged to LDS with
//                 global_load_async_to_lds_b128 (ASYNCcnt) when available.
// Fallback (small ws): single-pass kernel converting f32->bf16 while staging.
// ---------------------------------------------------------------------------

typedef __attribute__((ext_vector_type(16))) __bf16 v16bf;
typedef __attribute__((ext_vector_type(8)))  float  v8f;
typedef __attribute__((ext_vector_type(4)))  int    v4i;

#define NB 16384   // batch
#define ND 512     // input dim
#define NU 512     // units
#define NZ 2048    // 4*NU
#define NK 1024    // combined K (D + U)
#define KB 32      // K per WMMA step (bf16)
#define SA 40      // padded LDS row stride (bf16 elems); 80B rows stay 16B-aligned
#define BM 256     // rows per workgroup
#define NSTEP 32   // NK / KB

#if __has_builtin(__builtin_amdgcn_global_load_async_to_lds_b128)
#define HAVE_GASYNC 1
#else
#define HAVE_GASYNC 0
#endif

typedef __attribute__((address_space(1))) v4i v4i_g;   // global (AS1)
typedef __attribute__((address_space(3))) v4i v4i_l;   // LDS    (AS3)

__device__ __forceinline__ unsigned short f2bf(float f) {
  union { float f; unsigned u; } v; v.f = f;
  unsigned r = v.u + 0x7FFFu + ((v.u >> 16) & 1u);   // round-to-nearest-even
  return (unsigned short)(r >> 16);
}

#if __has_builtin(__builtin_amdgcn_cvt_pk_bf16_f32)
__device__ __forceinline__ unsigned pack2bf(float a, float b) {
  auto p = __builtin_amdgcn_cvt_pk_bf16_f32(a, b);
  return __builtin_bit_cast(unsigned, p);
}
#else
__device__ __forceinline__ unsigned pack2bf(float a, float b) {
  return (unsigned)f2bf(a) | ((unsigned)f2bf(b) << 16);
}
#endif

#if HAVE_GASYNC
__device__ __forceinline__ void cp_async16(const void* g, void* l) {
  __builtin_amdgcn_global_load_async_to_lds_b128((v4i_g*)g, (v4i_l*)l, 0, 0);
}
__device__ __forceinline__ void wait_async0() {
#if __has_builtin(__builtin_amdgcn_s_wait_asynccnt)
  __builtin_amdgcn_s_wait_asynccnt(0);
#else
  asm volatile("s_wait_asynccnt 0" ::: "memory");
#endif
}
#endif

// 16x32 bf16 WMMA fragment from an LDS row (ISA 7.12.2 layout):
// lanes 0-15 (lh=0): K=0..7 then 16..23 ; lanes 16-31 (lh=1): K=8..15, 24..31
__device__ __forceinline__ v16bf ld_frag(const unsigned short* row, int lh) {
  union { v16bf v; uint4 q[2]; } u;
  u.q[0] = *(const uint4*)(row + 8 * lh);
  u.q[1] = *(const uint4*)(row + 16 + 8 * lh);
  return u.v;
}

__device__ __forceinline__ float sigm(float x) {
  return 1.0f / (1.0f + __expf(-x));
}
__device__ __forceinline__ float tanh_fast(float x) {
  return 1.0f - 2.0f / (__expf(2.0f * x) + 1.0f);   // overflow-safe
}

// ---------------------------------------------------------------------------
// Prep: XH[row][k] = bf16( k<512 ? X[row][k] : H[row][k-512] )
// ---------------------------------------------------------------------------
__global__ __launch_bounds__(256) void pack_xh(const float* __restrict__ x,
                                               const float* __restrict__ h,
                                               unsigned short* __restrict__ XH) {
  const size_t i = ((size_t)blockIdx.x * 256 + threadIdx.x) * 8;   // elem idx
  const int row = (int)(i >> 10);
  const int k   = (int)(i & (NK - 1));
  const float* src = (k < ND) ? (x + (size_t)row * ND + k)
                              : (h + (size_t)row * NU + (k - ND));
  const float4 a = *(const float4*)src;
  const float4 b = *(const float4*)(src + 4);
  uint4 q;
  q.x = pack2bf(a.x, a.y); q.y = pack2bf(a.z, a.w);
  q.z = pack2bf(b.x, b.y); q.w = pack2bf(b.z, b.w);
  *(uint4*)(XH + i) = q;
}

// ---------------------------------------------------------------------------
// Prep: Wt[c][k] = bf16( k<512 ? W[k][c] : R[k-512][c] )   (transposed weights)
// ---------------------------------------------------------------------------
__global__ __launch_bounds__(256) void pack_wt(const float* __restrict__ W,
                                               const float* __restrict__ R,
                                               unsigned short* __restrict__ Wt) {
  const int t  = blockIdx.x * 256 + threadIdx.x;   // 0 .. NZ*NK/8-1
  const int c  = t & (NZ - 1);                     // adjacent lanes -> adjacent c
  const int k0 = (t >> 11) * 8;                    // t / NZ * 8
  float v[8];
#pragma unroll
  for (int i = 0; i < 8; ++i) {
    const int k = k0 + i;
    v[i] = (k < ND) ? W[(size_t)k * NZ + c] : R[(size_t)(k - ND) * NZ + c];
  }
  uint4 q;
  q.x = pack2bf(v[0], v[1]); q.y = pack2bf(v[2], v[3]);
  q.z = pack2bf(v[4], v[5]); q.w = pack2bf(v[6], v[7]);
  *(uint4*)(Wt + (size_t)c * NK + k0) = q;
}

// ---------------------------------------------------------------------------
// Main fused kernel (prepped bf16 inputs).
// WG: 256 thr = 8 waves; tile 256 rows x 32 u-cols (128 z-cols).
// Wave: 2 x 16-row strips, 4 gates x 2 N-tiles -> 16 accum tiles (128 VGPR).
// ---------------------------------------------------------------------------
__global__ __launch_bounds__(256) void lstm_wmma_pre(
    const unsigned short* __restrict__ XH,   // [NB][NK] bf16
    const unsigned short* __restrict__ Wt,   // [NZ][NK] bf16
    const float* __restrict__ cprev,
    const float* __restrict__ pw,
    const float* __restrict__ bias,
    float* __restrict__ out)
{
  const int tid  = threadIdx.x;
  const int lane = tid & 31;
  const int wv   = tid >> 5;
  const int ln   = lane & 15;
  const int lh   = lane >> 4;
  const int u0   = blockIdx.x * 32;
  const int m0   = blockIdx.y * BM;

  __shared__ __align__(16) unsigned short As [2][BM  * SA];
  __shared__ __align__(16) unsigned short Bsh[2][128 * SA];

  v8f acc[4][2][2];
#pragma unroll
  for (int t = 0; t < 2; ++t) {
    const int u = u0 + t * 16 + ln;
#pragma unroll
    for (int g = 0; g < 4; ++g) {
      const float bv = bias[g * NU + u];
      v8f a;
#pragma unroll
      for (int r = 0; r < 8; ++r) a[r] = bv;
      acc[g][t][0] = a;
      acc[g][t][1] = a;
    }
  }

  auto COMPUTE = [&](int pb) {
    v16bf afrag[2];
#pragma unroll
    for (int s = 0; s < 2; ++s)
      afrag[s] = ld_frag(&As[pb][(wv * 32 + s * 16 + ln) * SA], lh);
#pragma unroll
    for (int g = 0; g < 4; ++g) {
#pragma unroll
      for (int t = 0; t < 2; ++t) {
        const v16bf bfrag = ld_frag(&Bsh[pb][(g * 32 + t * 16 + ln) * SA], lh);
#pragma unroll
        for (int s = 0; s < 2; ++s)
          acc[g][t][s] = __builtin_amdgcn_wmma_f32_16x16x32_bf16(
              false, afrag[s], false, bfrag, (short)0, acc[g][t][s], false, false);
      }
    }
  };

#if HAVE_GASYNC
  // ---- async DMA staging: zero staging VGPRs, ASYNCcnt-tracked ----
  auto STAGE = [&](int ks, int buf) {
    const int k0 = ks * KB;
#pragma unroll
    for (int j = 0; j < 4; ++j) {              // A: 1024 x 16B chunks
      const int id = tid + 256 * j;
      const int m  = id >> 2;
      const int ko = (id & 3) * 8;
      cp_async16(XH + (((size_t)(m0 + m)) << 10) + k0 + ko,
                 &As[buf][m * SA + ko]);
    }
#pragma unroll
    for (int j = 0; j < 2; ++j) {              // B: 512 x 16B chunks
      const int id = tid + 256 * j;
      const int zc = id >> 2;                  // local z-col 0..127
      const int ko = (id & 3) * 8;
      const int wr = ((zc >> 5) << 9) + u0 + (zc & 31);   // gate*512 + u0 + n
      cp_async16(Wt + (((size_t)wr) << 10) + k0 + ko,
                 &Bsh[buf][zc * SA + ko]);
    }
  };

  STAGE(0, 0);
  wait_async0();
  __syncthreads();
  for (int ks = 0; ks < NSTEP; ++ks) {
    const int pb = ks & 1;
    if (ks + 1 < NSTEP) STAGE(ks + 1, pb ^ 1);   // DMA overlaps WMMA below
    COMPUTE(pb);
    if (ks + 1 < NSTEP) wait_async0();
    __syncthreads();
  }
#else
  // ---- register-staged fallback (bf16 raw copies, no conversion) ----
  uint4 rs[6];
  auto GLOADP = [&](int ks) {
    const int k0 = ks * KB;
#pragma unroll
    for (int j = 0; j < 4; ++j) {
      const int id = tid + 256 * j;
      const int m  = id >> 2;
      const int ko = (id & 3) * 8;
      rs[j] = *(const uint4*)(XH + (((size_t)(m0 + m)) << 10) + k0 + ko);
    }
#pragma unroll
    for (int j = 0; j < 2; ++j) {
      const int id = tid + 256 * j;
      const int zc = id >> 2;
      const int ko = (id & 3) * 8;
      const int wr = ((zc >> 5) << 9) + u0 + (zc & 31);
      rs[4 + j] = *(const uint4*)(Wt + (((size_t)wr) << 10) + k0 + ko);
    }
  };
  auto SSTOREP = [&](int buf) {
#pragma unroll
    for (int j = 0; j < 4; ++j) {
      const int id = tid + 256 * j;
      *(uint4*)(&As[buf][(id >> 2) * SA + (id & 3) * 8]) = rs[j];
    }
#pragma unroll
    for (int j = 0; j < 2; ++j) {
      const int id = tid + 256 * j;
      *(uint4*)(&Bsh[buf][(id >> 2) * SA + (id & 3) * 8]) = rs[4 + j];
    }
  };
  GLOADP(0); SSTOREP(0);
  __syncthreads();
  for (int ks = 0; ks < NSTEP; ++ks) {
    const int pb = ks & 1;
    if (ks + 1 < NSTEP) GLOADP(ks + 1);
    COMPUTE(pb);
    if (ks + 1 < NSTEP) SSTOREP(pb ^ 1);
    __syncthreads();
  }
#endif

  // ---- fused elementwise tail ----
#pragma unroll
  for (int s = 0; s < 2; ++s) {
    const int mbase = m0 + wv * 32 + s * 16 + lh * 8;
#pragma unroll
    for (int t = 0; t < 2; ++t) {
      const int u = u0 + t * 16 + ln;
      const float pwi = pw[u * 3 + 0];
      const float pwf = pw[u * 3 + 1];
      const float pwo = pw[u * 3 + 2];
#pragma unroll
      for (int r = 0; r < 8; ++r) {
        const int row = mbase + r;
        const float cp = cprev[(size_t)row * NU + u];
        const float ig = sigm(acc[0][t][s][r] + cp * pwi);
        const float fg = sigm(acc[1][t][s][r] + cp * pwf);
        const float gg = tanh_fast(acc[2][t][s][r]);
        const float cn = fg * cp + ig * gg;
        const float og = sigm(acc[3][t][s][r] + cn * pwo);
        const float hn = og * tanh_fast(cn);
        out[(size_t)row * NU + u] = hn;
        out[(size_t)NB * NU + (size_t)row * NU + u] = cn;
      }
    }
  }
}

// ---------------------------------------------------------------------------
// Fallback (ws too small): single pass, converts f32->bf16 while staging.
// ---------------------------------------------------------------------------
__global__ __launch_bounds__(256) void lstm_wmma_cvt(
    const float* __restrict__ x, const float* __restrict__ h,
    const float* __restrict__ cprev, const float* __restrict__ W,
    const float* __restrict__ R, const float* __restrict__ pw,
    const float* __restrict__ bias, float* __restrict__ out)
{
  const int tid  = threadIdx.x;
  const int lane = tid & 31;
  const int wv   = tid >> 5;
  const int ln   = lane & 15;
  const int lh   = lane >> 4;
  const int u0   = blockIdx.x * 32;
  const int m0   = blockIdx.y * BM;

  __shared__ __align__(16) unsigned short As [2][BM  * SA];
  __shared__ __align__(16) unsigned short Bsh[2][128 * SA];

  v8f acc[4][2][2];
#pragma unroll
  for (int t = 0; t < 2; ++t) {
    const int u = u0 + t * 16 + ln;
#pragma unroll
    for (int g = 0; g < 4; ++g) {
      const float bv = bias[g * NU + u];
      v8f a;
#pragma unroll
      for (int r = 0; r < 8; ++r) a[r] = bv;
      acc[g][t][0] = a;
      acc[g][t][1] = a;
    }
  }

  const int arow = tid >> 3;
  const int ak   = (tid & 7) * 4;
  const int bk   = tid >> 3;
  const int bc   = tid & 7;
  uint2 ra[8];
  uint2 rb[4];

  auto GLOAD = [&](int ks) {
    const float* __restrict__ As_ = (ks < 16) ? x : h;
    const float* __restrict__ Ws_ = (ks < 16) ? W : R;
    const int kc = (ks & 15) * KB;
#pragma unroll
    for (int j = 0; j < 8; ++j) {
      const float4 v = *(const float4*)(As_ + (size_t)(m0 + arow + 32 * j) * ND + kc + ak);
      ra[j].x = pack2bf(v.x, v.y);
      ra[j].y = pack2bf(v.z, v.w);
    }
#pragma unroll
    for (int j = 0; j < 4; ++j) {
      const int cc = (bc + 8 * j) * 4;
      const float4 v = *(const float4*)(Ws_ + (size_t)(kc + bk) * NZ + (cc >> 5) * NU + u0 + (cc & 31));
      rb[j].x = pack2bf(v.x, v.y);
      rb[j].y = pack2bf(v.z, v.w);
    }
  };
  auto SSTORE = [&](int buf) {
#pragma unroll
    for (int j = 0; j < 8; ++j)
      *(uint2*)(&As[buf][(arow + 32 * j) * SA + ak]) = ra[j];
#pragma unroll
    for (int j = 0; j < 4; ++j) {
      const int cc = (bc + 8 * j) * 4;
      unsigned short* p = &Bsh[buf][0];
      p[(cc + 0) * SA + bk] = (unsigned short)(rb[j].x);
      p[(cc + 1) * SA + bk] = (unsigned short)(rb[j].x >> 16);
      p[(cc + 2) * SA + bk] = (unsigned short)(rb[j].y);
      p[(cc + 3) * SA + bk] = (unsigned short)(rb[j].y >> 16);
    }
  };

  GLOAD(0); SSTORE(0);
  __syncthreads();
  for (int ks = 0; ks < NSTEP; ++ks) {
    const int pb = ks & 1;
    if (ks + 1 < NSTEP) GLOAD(ks + 1);
    v16bf afrag[2];
#pragma unroll
    for (int s = 0; s < 2; ++s)
      afrag[s] = ld_frag(&As[pb][(wv * 32 + s * 16 + ln) * SA], lh);
#pragma unroll
    for (int g = 0; g < 4; ++g) {
#pragma unroll
      for (int t = 0; t < 2; ++t) {
        const v16bf bfrag = ld_frag(&Bsh[pb][(g * 32 + t * 16 + ln) * SA], lh);
#pragma unroll
        for (int s = 0; s < 2; ++s)
          acc[g][t][s] = __builtin_amdgcn_wmma_f32_16x16x32_bf16(
              false, afrag[s], false, bfrag, (short)0, acc[g][t][s], false, false);
      }
    }
    if (ks + 1 < NSTEP) SSTORE(pb ^ 1);
    __syncthreads();
  }

#pragma unroll
  for (int s = 0; s < 2; ++s) {
    const int mbase = m0 + wv * 32 + s * 16 + lh * 8;
#pragma unroll
    for (int t = 0; t < 2; ++t) {
      const int u = u0 + t * 16 + ln;
      const float pwi = pw[u * 3 + 0];
      const float pwf = pw[u * 3 + 1];
      const float pwo = pw[u * 3 + 2];
#pragma unroll
      for (int r = 0; r < 8; ++r) {
        const int row = mbase + r;
        const float cp = cprev[(size_t)row * NU + u];
        const float ig = sigm(acc[0][t][s][r] + cp * pwi);
        const float fg = sigm(acc[1][t][s][r] + cp * pwf);
        const float gg = tanh_fast(acc[2][t][s][r]);
        const float cn = fg * cp + ig * gg;
        const float og = sigm(acc[3][t][s][r] + cn * pwo);
        const float hn = og * tanh_fast(cn);
        out[(size_t)row * NU + u] = hn;
        out[(size_t)NB * NU + (size_t)row * NU + u] = cn;
      }
    }
  }
}

extern "C" void kernel_launch(void* const* d_in, const int* in_sizes, int n_in,
                              void* d_out, int out_size, void* d_ws, size_t ws_size,
                              hipStream_t stream) {
  const float* x    = (const float*)d_in[0];
  const float* h    = (const float*)d_in[1];
  const float* c    = (const float*)d_in[2];
  const float* W    = (const float*)d_in[3];
  const float* R    = (const float*)d_in[4];
  const float* pw   = (const float*)d_in[5];
  const float* bias = (const float*)d_in[6];
  (void)in_sizes; (void)n_in; (void)out_size;

  const size_t XH_BYTES = (size_t)NB * NK * sizeof(unsigned short);  // 32 MB
  const size_t WT_BYTES = (size_t)NZ * NK * sizeof(unsigned short);  //  4 MB
  dim3 grid(NU / 32, NB / BM);   // 16 x 64

  if (ws_size >= XH_BYTES + WT_BYTES) {
    unsigned short* XH = (unsigned short*)d_ws;
    unsigned short* Wt = (unsigned short*)((char*)d_ws + XH_BYTES);
    pack_xh<<<(NB * NK / 8) / 256, 256, 0, stream>>>(x, h, XH);
    pack_wt<<<(NZ * NK / 8) / 256, 256, 0, stream>>>(W, R, Wt);
    lstm_wmma_pre<<<grid, 256, 0, stream>>>(XH, Wt, c, pw, bias, (float*)d_out);
  } else {
    lstm_wmma_cvt<<<grid, 256, 0, stream>>>(x, h, c, W, R, pw, bias, (float*)d_out);
  }
}
